// Transformer_40484361732746
// MI455X (gfx1250) — matrix operand order repaired and checked by reading, not compiled, not run on hardware
//
#include <hip/hip_runtime.h>
#include <hip/hip_bf16.h>
#include <math.h>

// ---------------------------------------------------------------------------
// Types / WMMA helpers (CDNA5 gfx1250, wave32)
// ---------------------------------------------------------------------------
typedef _Float16     v8h   __attribute__((ext_vector_type(8)));
typedef _Float16     v16h  __attribute__((ext_vector_type(16)));
typedef float        v8f   __attribute__((ext_vector_type(8)));
typedef unsigned int u32x4 __attribute__((ext_vector_type(4)));
typedef int          i32x4 __attribute__((ext_vector_type(4)));
typedef int          i32x8 __attribute__((ext_vector_type(8)));

#if __has_builtin(__builtin_amdgcn_tensor_load_to_lds) && \
    __has_builtin(__builtin_amdgcn_s_wait_tensorcnt)
#define HAVE_TDM 1
#else
#define HAVE_TDM 0
#endif

__device__ __forceinline__ v16h make_frag(v8h lo, v8h hi) {
    return __builtin_shufflevector(lo, hi, 0,1,2,3,4,5,6,7,8,9,10,11,12,13,14,15);
}

// A fragment (16x32 f16, M x K): lane L holds row m=L%16;
// halves h = A[m][klo+h], klo=(L/16)*16.
__device__ __forceinline__ v16h load_a_frag(const _Float16* __restrict__ row_k0, int lane) {
    int klo = (lane >> 4) << 4;
    v8h lo = *(const v8h*)(row_k0 + klo);
    v8h hi = *(const v8h*)(row_k0 + klo + 8);
    return make_frag(lo, hi);
}

// B fragment (32x16 f16, K x N) from column-contiguous storage (Bt row-major
// [N][K]): lane L holds col n=L%16; halves h = B[kb+h][n], kb=(L/16)*16.
__device__ __forceinline__ v16h load_b_frag(const _Float16* __restrict__ col_k0, int lane) {
    int kb = (lane >> 4) << 4;
    v8h lo = *(const v8h*)(col_k0 + kb);
    v8h hi = *(const v8h*)(col_k0 + kb + 8);
    return make_frag(lo, hi);
}

__device__ __forceinline__ v8f wmma32(v16h a, v16h b, v8f c) {
    return __builtin_amdgcn_wmma_f32_16x16x32_f16(false, a, false, b, (short)0, c, false, false);
}

__device__ __forceinline__ float gelu_f(float x) {
    return 0.5f * x * (1.0f + erff(x * 0.70710678118654752440f));
}

// ---------------------------------------------------------------------------
// DPP16 VALU reductions over each 16-lane half-wave (no LDS, no waits).
// Butterfly: quad_perm(1,0,3,2) [xor1], quad_perm(2,3,0,1) [xor2],
// row_half_mirror (0x141), row_mirror (0x140).
// ---------------------------------------------------------------------------
template<int CTRL>
__device__ __forceinline__ float dppf(float x) {
    return __int_as_float(
        __builtin_amdgcn_update_dpp(0, __float_as_int(x), CTRL, 0xF, 0xF, true));
}
__device__ __forceinline__ float red16_max(float x) {
    x = fmaxf(x, dppf<0xB1>(x));    // quad_perm [1,0,3,2]
    x = fmaxf(x, dppf<0x4E>(x));    // quad_perm [2,3,0,1]
    x = fmaxf(x, dppf<0x141>(x));   // row_half_mirror
    x = fmaxf(x, dppf<0x140>(x));   // row_mirror
    return x;
}
__device__ __forceinline__ float red16_sum(float x) {
    x += dppf<0xB1>(x);
    x += dppf<0x4E>(x);
    x += dppf<0x141>(x);
    x += dppf<0x140>(x);
    return x;
}

// ---------------------------------------------------------------------------
// Constants
// ---------------------------------------------------------------------------
#define Bc   8
#define Nc   1024
#define Dc   512
#define Hc   8
#define HDc  64
#define MLPc 2048
#define Mc   (Bc * Nc)       // 8192 rows
#define HHc  32
#define QK_SCALE 0.04419417382415921757f   // 512^-0.5

#define KCH       128                  // GEMM K staging chunk (elements)
#define LDSB_STR  136                  // padded LDS row stride in halfs (128+8)

// ---------------------------------------------------------------------------
// TDM: issue 2D tile load Global -> LDS (no wait). tile = tile_h rows x
// tile_w f16 elements, global row stride = row_stride elements. LDS rows
// padded via TDM pad feature: every 256B insert 16B (interval code 5,
// amount code 3) -> LDS row stride 136 halfs. D# layout per CDNA5 ISA ch.8.
// ---------------------------------------------------------------------------
__device__ __forceinline__ void tdm_issue_tile_f16(unsigned int lds_off_bytes,
                                                   const _Float16* gptr,
                                                   unsigned int tile_w,
                                                   unsigned int tile_h,
                                                   unsigned int row_stride)
{
#if HAVE_TDM
    unsigned long long ga = (unsigned long long)(const void*)gptr;
    u32x4 g0;
    g0[0] = 1u;                                        // count=1, user descriptor
    g0[1] = lds_off_bytes;                             // lds_addr
    g0[2] = (unsigned int)ga;                          // global_addr[31:0]
    g0[3] = (unsigned int)((ga >> 32) & 0x01FFFFFFu)   // global_addr[56:32]
            | (2u << 30);                              // type = 2 ("image")
    i32x8 g1;
    g1[0] = (int)((1u << 16)                           // data_size = 2 bytes
                  | (1u << 20)                         // pad_enable
                  | (5u << 22)                         // pad_interval: 256B
                  | (3u << 25));                       // pad_amount: 4 DWORDs
    g1[1] = (int)((tile_w & 0xFFFFu) << 16);           // tensor_dim0[15:0]
    g1[2] = (int)(((tile_w >> 16) & 0xFFFFu)           // tensor_dim0[31:16]
                  | ((tile_h & 0xFFFFu) << 16));       // tensor_dim1[15:0]
    g1[3] = (int)(((tile_h >> 16) & 0xFFFFu)           // tensor_dim1[31:16]
                  | ((tile_w & 0xFFFFu) << 16));       // tile_dim0
    g1[4] = (int)(tile_h & 0xFFFFu);                   // tile_dim1 (tile_dim2=0)
    g1[5] = (int)row_stride;                           // tensor_dim0_stride[31:0]
    g1[6] = 0;
    g1[7] = 0;
    i32x4 g2 = {0, 0, 0, 0};
    i32x4 g3 = {0, 0, 0, 0};
#if defined(__clang_major__) && (__clang_major__ >= 23)
    i32x8 gz = {0, 0, 0, 0, 0, 0, 0, 0};
    __builtin_amdgcn_tensor_load_to_lds(g0, g1, g2, g3, gz, 0);
#else
    __builtin_amdgcn_tensor_load_to_lds(g0, g1, g2, g3, 0);
#endif
#else
    (void)lds_off_bytes; (void)gptr; (void)tile_w; (void)tile_h; (void)row_stride;
#endif
}

// ---------------------------------------------------------------------------
// Generic WMMA GEMM:  C[M,N] = act(A[M,K](f16) x Bt[N,K](f16) + bias) (+resid)
// block = 256 (8 waves); each wave -> 16x64 tile; block -> 128x64.
// B panel (64 cols x KCH) staged to LDS via a double-buffered Tensor Data
// Mover pipeline: wave 0 issues chunk c+1 while all waves compute chunk c.
// ---------------------------------------------------------------------------
template<int ACT, int RESID, int OUTH>
__global__ void gemm_kernel(const _Float16* __restrict__ A,
                            const _Float16* __restrict__ Bt,
                            const float* __restrict__ bias,
                            const float* __restrict__ resid,
                            float* __restrict__ Cf,
                            _Float16* __restrict__ Ch,
                            int M, int N, int K)
{
    __shared__ _Float16 ldsB[2][64 * LDSB_STR];

    int lane = threadIdx.x & 31;
    int wave = threadIdx.x >> 5;
    int rowBase = blockIdx.y * 128 + wave * 16;
    int colBase = blockIdx.x * 64;

    v8f acc[4] = {};
    const _Float16* arow = A + (size_t)(rowBase + (lane & 15)) * K;
    int kb = (lane >> 4) << 4;
    const int nchunk = K / KCH;

#if HAVE_TDM
    if (wave == 0) {
        tdm_issue_tile_f16((unsigned int)(unsigned long long)(void*)&ldsB[0][0],
                           Bt + (size_t)colBase * K, KCH, 64, (unsigned int)K);
    }
    for (int c = 0; c < nchunk; ++c) {
        if (wave == 0) __builtin_amdgcn_s_wait_tensorcnt(0);
        __syncthreads();                       // publish chunk c to all waves
        if (wave == 0 && (c + 1) < nchunk) {   // overlap DMA of c+1 with compute
            tdm_issue_tile_f16(
                (unsigned int)(unsigned long long)(void*)&ldsB[(c + 1) & 1][0],
                Bt + (size_t)colBase * K + (size_t)(c + 1) * KCH,
                KCH, 64, (unsigned int)K);
        }
        const _Float16* bbase = &ldsB[c & 1][0];
        int kc = c * KCH;
        for (int k0 = 0; k0 < KCH; k0 += 32) {
            v16h a = load_a_frag(arow + kc + k0, lane);
#pragma unroll
            for (int t = 0; t < 4; ++t) {
                const _Float16* bp = bbase + (t * 16 + (lane & 15)) * LDSB_STR + k0;
                v8h lo = *(const v8h*)(bp + kb);
                v8h hi = *(const v8h*)(bp + kb + 8);
                acc[t] = wmma32(a, make_frag(lo, hi), acc[t]);
            }
        }
        __syncthreads();                       // all reads done before rewrite
    }
#else
    for (int c = 0; c < nchunk; ++c) {
        int kc = c * KCH;
        __syncthreads();
        for (int idx = threadIdx.x; idx < 64 * (KCH / 8); idx += 256) {
            int n  = idx / (KCH / 8);
            int c8 = (idx % (KCH / 8)) * 8;
            *(v8h*)&ldsB[0][n * LDSB_STR + c8] =
                *(const v8h*)(Bt + (size_t)(colBase + n) * K + kc + c8);
        }
        __syncthreads();
        for (int k0 = 0; k0 < KCH; k0 += 32) {
            v16h a = load_a_frag(arow + kc + k0, lane);
#pragma unroll
            for (int t = 0; t < 4; ++t) {
                const _Float16* bp = &ldsB[0][(t * 16 + (lane & 15)) * LDSB_STR + k0];
                v8h lo = *(const v8h*)(bp + kb);
                v8h hi = *(const v8h*)(bp + kb + 8);
                acc[t] = wmma32(a, make_frag(lo, hi), acc[t]);
            }
        }
    }
#endif

    int g = lane >> 4;
#pragma unroll
    for (int t = 0; t < 4; ++t) {
        int col = colBase + t * 16 + (lane & 15);
        float bv = bias ? bias[col] : 0.0f;
#pragma unroll
        for (int r = 0; r < 8; ++r) {
            int row = rowBase + r + 8 * g;
            float v = acc[t][r] + bv;
            if (ACT == 1) v = gelu_f(v);
            if (RESID)    v += resid[(size_t)row * N + col];
            if (OUTH) Ch[(size_t)row * N + col] = (_Float16)v;
            else      Cf[(size_t)row * N + col] = v;
        }
    }
}

// ---------------------------------------------------------------------------
// Flash attention: one wave per (b, h, 16-row tile). Online softmax over
// N=1024 in 32-column steps; row statistics via DPP16 VALU reductions.
// ---------------------------------------------------------------------------
__global__ void attn_kernel(const _Float16* __restrict__ Qs,
                            const _Float16* __restrict__ Kh,
                            const _Float16* __restrict__ Vt,
                            _Float16* __restrict__ Out)
{
    __shared__ _Float16 ldsP[16 * 32];

    int lane = threadIdx.x & 31;
    int mb = blockIdx.x;            // 0..63
    int h  = blockIdx.y;
    int b  = blockIdx.z;
    int m  = lane & 15;
    int g  = lane >> 4;
    int klo = g ? 8 : 0;

    size_t headOff = (size_t)(b * Hc + h) * Nc * HDc;
    int mrow = mb * 16 + m;

    const _Float16* qrow = Qs + headOff + (size_t)mrow * HDc;
    v16h qa0 = load_a_frag(qrow + 0,  lane);   // d = 0..31
    v16h qa1 = load_a_frag(qrow + 32, lane);   // d = 32..63

    v8f O[4] = {};
    float Mr[8], Lr[8];
#pragma unroll
    for (int r = 0; r < 8; ++r) { Mr[r] = -3.0e38f; Lr[r] = 0.0f; }

    for (int j0 = 0; j0 < Nc; j0 += 32) {
        // ---- S = Q K^T for 16 rows x 32 cols (two 16x16 tiles, K-dim 64)
        v8f s0 = {}, s1 = {};
        {
            const _Float16* kp0 = Kh + headOff + (size_t)(j0 + (lane & 15)) * HDc;
            s0 = wmma32(qa0, load_b_frag(kp0 + 0,  lane), s0);
            s0 = wmma32(qa1, load_b_frag(kp0 + 32, lane), s0);
            const _Float16* kp1 = Kh + headOff + (size_t)(j0 + 16 + (lane & 15)) * HDc;
            s1 = wmma32(qa0, load_b_frag(kp1 + 0,  lane), s1);
            s1 = wmma32(qa1, load_b_frag(kp1 + 32, lane), s1);
        }

        // ---- online softmax per C-layout row r (row m = r + 8*g, 16 lanes/row)
        float alpha[8];
#pragma unroll
        for (int r = 0; r < 8; ++r) {
            float rm = red16_max(fmaxf(s0[r], s1[r]));
            float mn = fmaxf(Mr[r], rm);
            alpha[r] = __expf(Mr[r] - mn);
            Mr[r] = mn;
            float p0 = __expf(s0[r] - mn);
            float p1 = __expf(s1[r] - mn);
            Lr[r] = Lr[r] * alpha[r] + red16_sum(p0 + p1);
            int prow = r + 8 * g;
            ldsP[prow * 32 + (lane & 15)]      = (_Float16)p0;
            ldsP[prow * 32 + 16 + (lane & 15)] = (_Float16)p1;
        }
#pragma unroll
        for (int t = 0; t < 4; ++t)
#pragma unroll
            for (int r = 0; r < 8; ++r)
                O[t][r] *= alpha[r];

        __syncthreads();
        // re-layout P (16x32) as an A fragment from LDS
        v16h pa;
        {
            const _Float16* pp = ldsP + m * 32;
            v8h lo = *(const v8h*)(pp + 16 * g);
            v8h hi = *(const v8h*)(pp + 16 * g + 8);
            pa = make_frag(lo, hi);
        }
        // ---- O += P * V  (V pre-transposed: Vt[(bh)*64 + d][n], n-contiguous)
#pragma unroll
        for (int t = 0; t < 4; ++t) {
            const _Float16* vp = Vt + ((size_t)(b * Hc + h) * HDc + t * 16 + (lane & 15)) * Nc + j0;
            O[t] = wmma32(pa, load_b_frag(vp, lane), O[t]);
        }
        __syncthreads();
    }

    // ---- normalize + write [b, row, h*64 + d] as f16
#pragma unroll
    for (int r = 0; r < 8; ++r) {
        float inv = 1.0f / Lr[r];
        int row = mb * 16 + r + 8 * g;
#pragma unroll
        for (int t = 0; t < 4; ++t) {
            size_t o = ((size_t)(b * Nc + row)) * Dc + h * HDc + t * 16 + (lane & 15);
            Out[o] = (_Float16)(O[t][r] * inv);
        }
    }
}

// ---------------------------------------------------------------------------
// LayerNorm: one block per row of 512 (256 threads, 2 elems each)
// ---------------------------------------------------------------------------
template<int OUTH>
__global__ void ln_kernel(const float* __restrict__ X,
                          const float* __restrict__ gg,
                          const float* __restrict__ bb,
                          float* __restrict__ Yf,
                          _Float16* __restrict__ Yh)
{
    __shared__ float red[256];
    int row = blockIdx.x, tid = threadIdx.x;
    const float* xr = X + (size_t)row * Dc;
    float a0 = xr[tid], a1 = xr[tid + 256];

    red[tid] = a0 + a1;
    __syncthreads();
    for (int st = 128; st > 0; st >>= 1) {
        if (tid < st) red[tid] += red[tid + st];
        __syncthreads();
    }
    float mean = red[0] * (1.0f / 512.0f);
    __syncthreads();

    float d0 = a0 - mean, d1 = a1 - mean;
    red[tid] = d0 * d0 + d1 * d1;
    __syncthreads();
    for (int st = 128; st > 0; st >>= 1) {
        if (tid < st) red[tid] += red[tid + st];
        __syncthreads();
    }
    float rstd = rsqrtf(red[0] * (1.0f / 512.0f) + 1e-5f);

    float y0 = d0 * rstd * gg[tid]       + bb[tid];
    float y1 = d1 * rstd * gg[tid + 256] + bb[tid + 256];
    if (OUTH) {
        Yh[(size_t)row * Dc + tid]       = (_Float16)y0;
        Yh[(size_t)row * Dc + tid + 256] = (_Float16)y1;
    } else {
        Yf[(size_t)row * Dc + tid]       = y0;
        Yf[(size_t)row * Dc + tid + 256] = y1;
    }
}

// ---------------------------------------------------------------------------
// Small utility kernels
// ---------------------------------------------------------------------------
__global__ void init_kernel(const float* __restrict__ in, float* __restrict__ x,
                            float* __restrict__ res)
{
    size_t i = (size_t)blockIdx.x * blockDim.x + threadIdx.x;
    if (i >= (size_t)Mc * Dc) return;
    float v = in[i];
    x[i] = v; res[i] = v;
}

__global__ void resmerge_kernel(float* __restrict__ x, float* __restrict__ res)
{
    size_t i = (size_t)blockIdx.x * blockDim.x + threadIdx.x;
    if (i >= (size_t)Mc * Dc) return;
    float v = x[i] + res[i];
    x[i] = v; res[i] = v;
}

// fp32 [K][N] -> f16 transposed [N][K]
__global__ void wt_kernel(const float* __restrict__ W, _Float16* __restrict__ Wt,
                          int K, int N)
{
    size_t i = (size_t)blockIdx.x * blockDim.x + threadIdx.x;
    if (i >= (size_t)K * N) return;
    int k = (int)(i / N), n = (int)(i % N);
    Wt[(size_t)n * K + k] = (_Float16)W[i];
}

// qkv f16 [M, 3*512] -> Q (scaled) [B,H,N,64], K [B,H,N,64], V^T [B,H,64,N]
__global__ void split_kernel(const _Float16* __restrict__ qkv,
                             _Float16* __restrict__ Qs,
                             _Float16* __restrict__ Kh,
                             _Float16* __restrict__ Vt)
{
    size_t i = (size_t)blockIdx.x * blockDim.x + threadIdx.x;
    if (i >= (size_t)Mc * 3 * Dc) return;
    int mtok = (int)(i / (3 * Dc));
    int c    = (int)(i % (3 * Dc));
    int b = mtok >> 10, n = mtok & 1023;
    int part = c / Dc, w = c % Dc;
    int h = w >> 6, d = w & 63;
    _Float16 v = qkv[i];
    size_t rowOff = (((size_t)(b * Hc + h)) * Nc + n) * HDc + d;
    if (part == 0)      Qs[rowOff] = (_Float16)((float)v * QK_SCALE);
    else if (part == 1) Kh[rowOff] = v;
    else Vt[(((size_t)(b * Hc + h)) * HDc + d) * Nc + n] = v;
}

// avg & max pool over tokens: xhat [B,N,512] -> [B,512]
__global__ void pool_kernel(const float* __restrict__ X,
                            float* __restrict__ avg, float* __restrict__ mx)
{
    int i = blockIdx.x * blockDim.x + threadIdx.x;
    if (i >= Bc * Dc) return;
    int b = i >> 9, c = i & 511;
    float s = 0.0f, m = -3.0e38f;
    for (int n = 0; n < Nc; ++n) {
        float v = X[((size_t)b * Nc + n) * Dc + c];
        s += v; m = fmaxf(m, v);
    }
    avg[i] = s * (1.0f / (float)Nc);
    mx[i]  = m;
}

// SE gate: scale[b,c] = sigmoid(mlp(avg)+mlp(mx)) ; one block per batch
__global__ void se_kernel(const float* __restrict__ avg, const float* __restrict__ mx,
                          const float* __restrict__ w1, const float* __restrict__ b1,
                          const float* __restrict__ w2, const float* __restrict__ b2,
                          float* __restrict__ scale)
{
    __shared__ float ha[32], hm[32];
    int b = blockIdx.x, tid = threadIdx.x;
    if (tid < 32) {
        float sa = b1[tid], sm = b1[tid];
        for (int c = 0; c < Dc; ++c) {
            float w = w1[(size_t)c * 32 + tid];
            sa += avg[b * Dc + c] * w;
            sm += mx[b * Dc + c] * w;
        }
        ha[tid] = fmaxf(sa, 0.0f);
        hm[tid] = fmaxf(sm, 0.0f);
    }
    __syncthreads();
    for (int c = tid; c < Dc; c += 256) {
        float s = 2.0f * b2[c];
        for (int r = 0; r < 32; ++r)
            s += (ha[r] + hm[r]) * w2[(size_t)r * Dc + c];
        scale[b * Dc + c] = 1.0f / (1.0f + __expf(-s));
    }
}

// xhat * SE scale -> f16 A operand for w1 GEMM
__global__ void scalemul_kernel(const float* __restrict__ X,
                                const float* __restrict__ scale,
                                _Float16* __restrict__ Y)
{
    size_t i = (size_t)blockIdx.x * blockDim.x + threadIdx.x;
    if (i >= (size_t)Mc * Dc) return;
    int m = (int)(i >> 9), c = (int)(i & 511);
    int b = m >> 10;
    Y[i] = (_Float16)(X[i] * scale[(b << 9) + c]);
}

// depthwise 3x3 conv on 32x32 token grid, + bias, gelu; f16 in/out [M, MLP]
__global__ void dwconv_kernel(const _Float16* __restrict__ H,
                              const float* __restrict__ Wd,
                              const float* __restrict__ Bd,
                              _Float16* __restrict__ H2)
{
    size_t i = (size_t)blockIdx.x * blockDim.x + threadIdx.x;
    if (i >= (size_t)Mc * MLPc) return;
    int m = (int)(i / MLPc), c = (int)(i % MLPc);
    int b = m >> 10, n = m & 1023;
    int y = n >> 5, x = n & 31;
    float acc = 0.0f;
#pragma unroll
    for (int dy = 0; dy < 3; ++dy) {
        int yy = y + dy - 1;
        if (yy < 0 || yy >= HHc) continue;
#pragma unroll
        for (int dx = 0; dx < 3; ++dx) {
            int xx = x + dx - 1;
            if (xx < 0 || xx >= HHc) continue;
            acc += (float)H[((size_t)(b * Nc + yy * HHc + xx)) * MLPc + c]
                   * Wd[(size_t)c * 9 + dy * 3 + dx];
        }
    }
    H2[i] = (_Float16)gelu_f(acc + Bd[c]);
}

// ---------------------------------------------------------------------------
// Host driver
// ---------------------------------------------------------------------------
extern "C" void kernel_launch(void* const* d_in, const int* in_sizes, int n_in,
                              void* d_out, int out_size, void* d_ws, size_t ws_size,
                              hipStream_t stream)
{
    (void)in_sizes; (void)n_in; (void)out_size; (void)ws_size;

    const float* in_x  = (const float*)d_in[0];
    const float* ln1_g = (const float*)d_in[1];
    const float* ln1_b = (const float*)d_in[2];
    const float* w_qkv = (const float*)d_in[3];
    const float* w_out = (const float*)d_in[4];
    const float* b_out = (const float*)d_in[5];
    const float* ln2_g = (const float*)d_in[6];
    const float* ln2_b = (const float*)d_in[7];
    const float* se_w1 = (const float*)d_in[8];
    const float* se_b1 = (const float*)d_in[9];
    const float* se_w2 = (const float*)d_in[10];
    const float* se_b2 = (const float*)d_in[11];
    const float* w1    = (const float*)d_in[12];
    const float* b1    = (const float*)d_in[13];
    const float* dw_w  = (const float*)d_in[14];
    const float* dw_b  = (const float*)d_in[15];
    const float* w2    = (const float*)d_in[16];
    const float* b2    = (const float*)d_in[17];

    char* ws = (char*)d_ws;
    size_t off = 0;
    auto alloc = [&](size_t bytes) -> void* {
        void* p = ws + off;
        off += (bytes + 255) & ~(size_t)255;
        return p;
    };

    float*     x_buf = (float*)d_out;                                  // running x
    float*     res   = (float*)alloc((size_t)Mc * Dc * 4);
    _Float16*  lnA   = (_Float16*)alloc((size_t)Mc * Dc * 2);
    _Float16*  qkvh  = (_Float16*)alloc((size_t)Mc * 3 * Dc * 2);
    _Float16*  Qs    = (_Float16*)alloc((size_t)Bc * Hc * Nc * HDc * 2);
    _Float16*  Kh    = (_Float16*)alloc((size_t)Bc * Hc * Nc * HDc * 2);
    _Float16*  Vt    = (_Float16*)alloc((size_t)Bc * Hc * Nc * HDc * 2);
    _Float16*  attnA = (_Float16*)alloc((size_t)Mc * Dc * 2);
    float*     xhat  = (float*)alloc((size_t)Mc * Dc * 4);
    float*     avgp  = (float*)alloc((size_t)Bc * Dc * 4);
    float*     mxp   = (float*)alloc((size_t)Bc * Dc * 4);
    float*     sescl = (float*)alloc((size_t)Bc * Dc * 4);
    _Float16*  w1A   = (_Float16*)alloc((size_t)Mc * Dc * 2);
    _Float16*  hbuf  = (_Float16*)alloc((size_t)Mc * MLPc * 2);
    _Float16*  h2buf = (_Float16*)alloc((size_t)Mc * MLPc * 2);
    _Float16*  wT    = (_Float16*)alloc((size_t)MLPc * Dc * 2);        // reused

    const int T = 256;
    init_kernel<<<(Mc * Dc + T - 1) / T, T, 0, stream>>>(in_x, x_buf, res);

    for (int i = 0; i < 4; ++i) {
        const float* l1g = ln1_g + (size_t)i * Dc;
        const float* l1b = ln1_b + (size_t)i * Dc;
        const float* wq  = w_qkv + (size_t)i * Dc * 3 * Dc;
        const float* wo  = w_out + (size_t)i * Dc * Dc;
        const float* bo  = b_out + (size_t)i * Dc;
        const float* l2g = ln2_g + (size_t)i * Dc;
        const float* l2b = ln2_b + (size_t)i * Dc;
        const float* sw1 = se_w1 + (size_t)i * Dc * 32;
        const float* sb1 = se_b1 + (size_t)i * 32;
        const float* sw2 = se_w2 + (size_t)i * 32 * Dc;
        const float* sb2 = se_b2 + (size_t)i * Dc;
        const float* w1i = w1    + (size_t)i * Dc * MLPc;
        const float* b1i = b1    + (size_t)i * MLPc;
        const float* dwi = dw_w  + (size_t)i * MLPc * 9;
        const float* dbi = dw_b  + (size_t)i * MLPc;
        const float* w2i = w2    + (size_t)i * MLPc * Dc;
        const float* b2i = b2    + (size_t)i * Dc;

        // ---- attention branch
        ln_kernel<1><<<Mc, 256, 0, stream>>>(x_buf, l1g, l1b, nullptr, lnA);
        wt_kernel<<<(Dc * 3 * Dc + T - 1) / T, T, 0, stream>>>(wq, wT, Dc, 3 * Dc);
        gemm_kernel<0, 0, 1><<<dim3(3 * Dc / 64, Mc / 128), 256, 0, stream>>>(
            lnA, wT, nullptr, nullptr, nullptr, qkvh, Mc, 3 * Dc, Dc);
        split_kernel<<<((size_t)Mc * 3 * Dc + T - 1) / T, T, 0, stream>>>(qkvh, Qs, Kh, Vt);
        attn_kernel<<<dim3(Nc / 16, Hc, Bc), 32, 0, stream>>>(Qs, Kh, Vt, attnA);
        wt_kernel<<<(Dc * Dc + T - 1) / T, T, 0, stream>>>(wo, wT, Dc, Dc);
        gemm_kernel<0, 1, 0><<<dim3(Dc / 64, Mc / 128), 256, 0, stream>>>(
            attnA, wT, bo, x_buf, x_buf, nullptr, Mc, Dc, Dc);

        // ---- SE + MLP branch
        ln_kernel<0><<<Mc, 256, 0, stream>>>(x_buf, l2g, l2b, xhat, nullptr);
        pool_kernel<<<(Bc * Dc + T - 1) / T, T, 0, stream>>>(xhat, avgp, mxp);
        se_kernel<<<Bc, 256, 0, stream>>>(avgp, mxp, sw1, sb1, sw2, sb2, sescl);
        scalemul_kernel<<<((size_t)Mc * Dc + T - 1) / T, T, 0, stream>>>(xhat, sescl, w1A);
        wt_kernel<<<(Dc * MLPc + T - 1) / T, T, 0, stream>>>(w1i, wT, Dc, MLPc);
        gemm_kernel<1, 0, 1><<<dim3(MLPc / 64, Mc / 128), 256, 0, stream>>>(
            w1A, wT, b1i, nullptr, nullptr, hbuf, Mc, MLPc, Dc);
        dwconv_kernel<<<((size_t)Mc * MLPc + T - 1) / T, T, 0, stream>>>(hbuf, dwi, dbi, h2buf);
        wt_kernel<<<(MLPc * Dc + T - 1) / T, T, 0, stream>>>(w2i, wT, MLPc, Dc);
        gemm_kernel<1, 1, 0><<<dim3(Dc / 64, Mc / 128), 256, 0, stream>>>(
            h2buf, wT, b2i, x_buf, x_buf, nullptr, Mc, Dc, MLPc);

        if ((i & 1) == 1)
            resmerge_kernel<<<(Mc * Dc + T - 1) / T, T, 0, stream>>>(x_buf, res);
    }
}
